// GATConv_48387101557182
// MI455X (gfx1250) — matrix operand order repaired and checked by reading, not compiled, not run on hardware
//
#include <hip/hip_runtime.h>
#include <math.h>

typedef __bf16 bf16_t;
typedef __attribute__((ext_vector_type(16))) __bf16 v16bf;
typedef __attribute__((ext_vector_type(8)))  __bf16 v8bf;
typedef __attribute__((ext_vector_type(8)))  float  v8f;
typedef __attribute__((ext_vector_type(4)))  float  v4f;
typedef __attribute__((__vector_size__(16))) int    v4i_t;

#define NROW 8192
#define FDIM 512
#define FH   1024

#if __has_builtin(__builtin_amdgcn_global_load_async_to_lds_b128) && \
    __has_builtin(__builtin_amdgcn_s_wait_asynccnt)
#define ASYNC_LDS 1
#endif

// ---------------------------------------------------------------------------
// helpers
// ---------------------------------------------------------------------------
__device__ __forceinline__ v8f v8f_zero() {
  v8f r;
#pragma unroll
  for (int i = 0; i < 8; ++i) r[i] = 0.0f;
  return r;
}

__device__ __forceinline__ v8f wmma_bf16(v16bf a, v16bf b, v8f c) {
  return __builtin_amdgcn_wmma_f32_16x16x32_bf16(false, a, false, b,
                                                 (short)0, c, false, false);
}

// A fragment (16x32, M x K), source row-major M[row][k].
// lanes 0-15: row=lane, K {0..7,16..23}; lanes 16-31: K {8..15,24..31}.
__device__ __forceinline__ v16bf load_a16(const bf16_t* base, int ld,
                                          int row0, int k0) {
  int lane = threadIdx.x & 31;
  int m = lane & 15;
  int hi = lane >> 4;
  const bf16_t* p = base + (size_t)(row0 + m) * (size_t)ld + k0;
  v8bf lo = *(const v8bf*)(p + hi * 8);
  v8bf hh = *(const v8bf*)(p + 16 + hi * 8);
  v16bf r;
#pragma unroll
  for (int i = 0; i < 8; ++i) { r[i] = lo[i]; r[8 + i] = hh[i]; }
  return r;
}

// B fragment (32x16, K x N): B[k][n] = src[(n0+n)*ld + k0+k].
// lanes 0-15: n=lane, K {0..7,16..23}; lanes 16-31: K {8..15,24..31}. ld,k0 %16==0.
__device__ __forceinline__ v16bf load_b16(const bf16_t* base, int ld,
                                          int n0, int k0) {
  int lane = threadIdx.x & 31;
  int n = lane & 15;
  int hi = lane >> 4;
  const bf16_t* p = base + (size_t)(n0 + n) * (size_t)ld + k0;
  v8bf lo = *(const v8bf*)(p + hi * 8);
  v8bf hh = *(const v8bf*)(p + 16 + hi * 8);
  v16bf r;
#pragma unroll
  for (int i = 0; i < 8; ++i) { r[i] = lo[i]; r[8 + i] = hh[i]; }
  return r;
}

// 16-byte global -> LDS copy: async path (ASYNCcnt) when available.
__device__ __forceinline__ void cp16(bf16_t* dst_lds, const bf16_t* src_g) {
#ifdef ASYNC_LDS
  __builtin_amdgcn_global_load_async_to_lds_b128(
      (__attribute__((address_space(1))) v4i_t*)src_g,
      (__attribute__((address_space(3))) v4i_t*)dst_lds, 0, 0);
#else
  *(v8bf*)dst_lds = *(const v8bf*)src_g;
#endif
}
__device__ __forceinline__ void cp_wait() {
#ifdef ASYNC_LDS
  __builtin_amdgcn_s_wait_asynccnt(0);
#endif
}

// ---------------------------------------------------------------------------
// K1: WhWs = X @ W ; T = tanh(Wh + Ws + bias). Emits Wh, WhT, T (bf16).
// Grid (NROW/128, FDIM/64), 256 threads (8 waves: 4 M x {Wh,Ws}).
// ---------------------------------------------------------------------------
__global__ __launch_bounds__(256) void k1_gemm_tanh(
    const float* __restrict__ X, const float* __restrict__ W,
    const float* __restrict__ a_w, const float* __restrict__ a_b,
    bf16_t* __restrict__ Whb, bf16_t* __restrict__ WhT,
    bf16_t* __restrict__ Tb)
{
  __shared__ __align__(32) bf16_t Xs[128 * 48];
  __shared__ __align__(32) bf16_t Wt[128 * 48];
  __shared__ float Ss[128 * 64];

  const int tid  = threadIdx.x;
  const int wave = tid >> 5;
  const int lane = tid & 31;
  const int lm = lane & 15, lhi = lane >> 4;
  const int mw = wave & 3;
  const int nh = wave >> 2;
  const int row0 = blockIdx.x * 128;
  const int c0   = blockIdx.y * 64;

  v8f acc[2][4];
#pragma unroll
  for (int mi = 0; mi < 2; ++mi)
#pragma unroll
    for (int ni = 0; ni < 4; ++ni) acc[mi][ni] = v8f_zero();

  for (int k0 = 0; k0 < FDIM; k0 += 32) {
    {
      int r = tid >> 1, h = tid & 1;
      const v4f* src = (const v4f*)(X + (size_t)(row0 + r) * FDIM + k0 + h * 16);
      bf16_t* dst = Xs + r * 48 + h * 16;
#pragma unroll
      for (int q = 0; q < 4; ++q) {
        v4f x = src[q];
#pragma unroll
        for (int e = 0; e < 4; ++e) dst[q * 4 + e] = (bf16_t)x[e];
      }
    }
    {
      int kk = tid & 31;
      int cb = (tid >> 5) * 16;
      int gcol = (cb < 64) ? (c0 + cb) : (512 + c0 + (cb - 64));
      const v4f* src = (const v4f*)(W + (size_t)(k0 + kk) * FH + gcol);
#pragma unroll
      for (int q = 0; q < 4; ++q) {
        v4f x = src[q];
#pragma unroll
        for (int e = 0; e < 4; ++e) Wt[(cb + q * 4 + e) * 48 + kk] = (bf16_t)x[e];
      }
    }
    __syncthreads();
    v16bf bfr[4];
#pragma unroll
    for (int ni = 0; ni < 4; ++ni) bfr[ni] = load_b16(Wt, 48, nh * 64 + ni * 16, 0);
#pragma unroll
    for (int mi = 0; mi < 2; ++mi) {
      v16bf a = load_a16(Xs, 48, mw * 32 + mi * 16, 0);
#pragma unroll
      for (int ni = 0; ni < 4; ++ni) acc[mi][ni] = wmma_bf16(a, bfr[ni], acc[mi][ni]);
    }
    __syncthreads();
  }

  if (nh == 1) {
#pragma unroll
    for (int mi = 0; mi < 2; ++mi)
#pragma unroll
      for (int ni = 0; ni < 4; ++ni)
#pragma unroll
        for (int v = 0; v < 8; ++v) {
          int r = mw * 32 + mi * 16 + v + 8 * lhi;
          Ss[r * 64 + ni * 16 + lm] = acc[mi][ni][v];
        }
  }
  __syncthreads();
  if (nh == 0) {
    float bias = a_w[0] + a_b[0];
#pragma unroll
    for (int mi = 0; mi < 2; ++mi)
#pragma unroll
      for (int ni = 0; ni < 4; ++ni)
#pragma unroll
        for (int v = 0; v < 8; ++v) {
          int r  = mw * 32 + mi * 16 + v + 8 * lhi;
          int cc = ni * 16 + lm;
          float wh = acc[mi][ni][v];
          float ws = Ss[r * 64 + cc];
          float t  = tanhf(wh + ws + bias);
          int grow = row0 + r, gcol = c0 + cc;
          Whb[(size_t)grow * FDIM + gcol] = (bf16_t)wh;
          WhT[(size_t)gcol * NROW + grow] = (bf16_t)wh;
          Tb[(size_t)grow * FDIM + gcol]  = (bf16_t)t;
        }
  }
}

// ---------------------------------------------------------------------------
// K2: flash attention, M_TILE = 64 rows / WG, 256 threads (8 waves).
// S: wave (mw 0..3, nw 0..1) computes one 16x16 tile with full K=512 from LDS.
// PV: each wave owns a 64x64 O slice (fw = wave*64), B from L2-resident WhT.
// Wh j-tiles are double-buffered: the async DMA for block jb+32 overlaps the
// whole S + softmax + PV of block jb (ASYNCcnt waited only at iteration tail).
// Dynamic LDS: Tt[64][528] + 2x Wj[32][528] + Pt[64][48] + stats (~140 KB).
// ---------------------------------------------------------------------------
#define TT_LD 528
#define WJ_LD 528
#define PT_LD 48
#define K2_SMEM_BYTES (((64 * TT_LD + 64 * WJ_LD) * 2) + (64 * PT_LD * 2) + 448 * 4)

__device__ __forceinline__ void stage_wj(bf16_t* dst_base,
                                         const bf16_t* __restrict__ Whb,
                                         int jb, int tid) {
  int r  = tid >> 3;
  int cb = (tid & 7) * 64;
  const bf16_t* src = Whb + (size_t)(jb + r) * FDIM + cb;
  bf16_t* dst = dst_base + r * WJ_LD + cb;
#pragma unroll
  for (int i = 0; i < 8; ++i) cp16(dst + i * 8, src + i * 8);
}

__global__ __launch_bounds__(256) void k2_attention(
    const bf16_t* __restrict__ Whb,  // [NROW,FDIM]
    const bf16_t* __restrict__ WhT,  // [FDIM,NROW]
    const bf16_t* __restrict__ Tb,   // [NROW,FDIM]
    const float* __restrict__ X,     // [NROW,FDIM]
    float* __restrict__ Y)           // [NROW,FDIM]
{
  extern __shared__ char smem[];
  bf16_t* Tt  = (bf16_t*)smem;          // [64][528]
  bf16_t* Wj0 = Tt + 64 * TT_LD;        // [32][528]
  bf16_t* Wj1 = Wj0 + 32 * WJ_LD;       // [32][528]
  bf16_t* Pt  = Wj1 + 32 * WJ_LD;       // [64][48]
  float* fbase = (float*)(Pt + 64 * PT_LD);
  float* m_s   = fbase;        // [64]
  float* l_s   = fbase + 64;   // [64]
  float* al_s  = fbase + 128;  // [64]
  float* rmax2 = fbase + 192;  // [2][64]
  float* rsum2 = fbase + 320;  // [2][64]

  const int tid  = threadIdx.x;
  const int wave = tid >> 5;
  const int lane = tid & 31;
  const int lm = lane & 15, lhi = lane >> 4;
  const int mw = wave & 3;   // S row tile (16 rows)
  const int nw = wave >> 2;  // S col tile (16 of 32)
  const int fw = wave * 64;  // O feature slice
  const int row0 = blockIdx.x * 64;

  // stage T tile [64][512] + first Wh j-tile (both async)
  {
    int r  = tid >> 2;
    int cb = (tid & 3) * 128;
    const bf16_t* src = Tb + (size_t)(row0 + r) * FDIM + cb;
    bf16_t* dst = Tt + r * TT_LD + cb;
#pragma unroll
    for (int i = 0; i < 16; ++i) cp16(dst + i * 8, src + i * 8);
  }
  stage_wj(Wj0, Whb, 0, tid);
  if (tid < 64) { m_s[tid] = -3.0e38f; l_s[tid] = 0.0f; }

  v8f o[4][4];
#pragma unroll
  for (int mi = 0; mi < 4; ++mi)
#pragma unroll
    for (int ni = 0; ni < 4; ++ni) o[mi][ni] = v8f_zero();

  cp_wait();
  __syncthreads();

  for (int jb = 0; jb < NROW; jb += 32) {
    int buf = (jb >> 5) & 1;
    bf16_t* Wjc = buf ? Wj1 : Wj0;
    bf16_t* Wjn = buf ? Wj0 : Wj1;

    // prefetch next j-tile: DMA runs under the entire block's compute
    if (jb + 32 < NROW) stage_wj(Wjn, Whb, jb + 32, tid);

    // ---- S tile (16x16 per wave, K = 512) ----
    v8f c = v8f_zero();
#pragma unroll
    for (int kk = 0; kk < 16; ++kk) {
      v16bf a = load_a16(Tt, TT_LD, mw * 16, kk * 32);
      v16bf b = load_b16(Wjc, WJ_LD, nw * 16, kk * 32);
      c = wmma_bf16(a, b, c);
    }

    // ---- per-row max of this 16-col tile ----
    {
      float rmx[8];
#pragma unroll
      for (int v = 0; v < 8; ++v) {
        float x = c[v];
        x = fmaxf(x, __shfl_xor(x, 1));
        x = fmaxf(x, __shfl_xor(x, 2));
        x = fmaxf(x, __shfl_xor(x, 4));
        x = fmaxf(x, __shfl_xor(x, 8));
        rmx[v] = x;
      }
      if (lm == 0) {
#pragma unroll
        for (int v = 0; v < 8; ++v)
          rmax2[nw * 64 + mw * 16 + v + 8 * lhi] = rmx[v];
      }
    }
    __syncthreads();

    // ---- online stats ----
    if (tid < 64) {
      float mt = fmaxf(rmax2[tid], rmax2[64 + tid]);
      float mo = m_s[tid];
      float mn = fmaxf(mo, mt);
      al_s[tid] = __expf(mo - mn);
      m_s[tid]  = mn;
    }
    __syncthreads();

    // ---- P = exp(S - m) -> LDS + row partial sums ----
    {
      float rs[8];
#pragma unroll
      for (int v = 0; v < 8; ++v) {
        int R = mw * 16 + v + 8 * lhi;
        float p = __expf(c[v] - m_s[R]);
        Pt[R * PT_LD + nw * 16 + lm] = (bf16_t)p;
        float s = p;
        s += __shfl_xor(s, 1);
        s += __shfl_xor(s, 2);
        s += __shfl_xor(s, 4);
        s += __shfl_xor(s, 8);
        rs[v] = s;
      }
      if (lm == 0) {
#pragma unroll
        for (int v = 0; v < 8; ++v)
          rsum2[nw * 64 + mw * 16 + v + 8 * lhi] = rs[v];
      }
    }
    __syncthreads();

    if (tid < 64) l_s[tid] = l_s[tid] * al_s[tid] + rsum2[tid] + rsum2[64 + tid];

    // ---- rescale O, accumulate P @ Wh ----
#pragma unroll
    for (int mi = 0; mi < 4; ++mi) {
      float al[8];
#pragma unroll
      for (int v = 0; v < 8; ++v) al[v] = al_s[mi * 16 + v + 8 * lhi];
      v16bf ap = load_a16(Pt, PT_LD, mi * 16, 0);
#pragma unroll
      for (int ni = 0; ni < 4; ++ni) {
#pragma unroll
        for (int v = 0; v < 8; ++v) o[mi][ni][v] = o[mi][ni][v] * al[v];
        v16bf b = load_b16(WhT, NROW, fw + ni * 16, jb);
        o[mi][ni] = wmma_bf16(ap, b, o[mi][ni]);
      }
    }

    cp_wait();        // prefetched j-tile has landed
    __syncthreads();  // ... and is visible to all waves
  }

  // ---- epilogue: O/l + residual ----
#pragma unroll
  for (int mi = 0; mi < 4; ++mi) {
    float li[8];
#pragma unroll
    for (int v = 0; v < 8; ++v) li[v] = l_s[mi * 16 + v + 8 * lhi];
#pragma unroll
    for (int ni = 0; ni < 4; ++ni)
#pragma unroll
      for (int v = 0; v < 8; ++v) {
        int r = mi * 16 + v + 8 * lhi;
        int grow = row0 + r;
        int gcol = fw + ni * 16 + lm;
        float val = o[mi][ni][v] / li[v] + X[(size_t)grow * FDIM + gcol];
        Y[(size_t)grow * FDIM + gcol] = val;
      }
  }
}

// ---------------------------------------------------------------------------
// K3: per-feature mean / rstd over nodes
// ---------------------------------------------------------------------------
__global__ __launch_bounds__(256) void k3_bnstats(const float* __restrict__ Y,
                                                  float* __restrict__ stats) {
  __shared__ float s1[4][64], s2[4][64];
  int tid = threadIdx.x;
  int cf = tid & 63, rs = tid >> 6;
  int col = blockIdx.x * 64 + cf;
  float a = 0.0f, b = 0.0f;
  for (int r = rs; r < NROW; r += 4) {
    float v = Y[(size_t)r * FDIM + col];
    a += v;
    b += v * v;
  }
  s1[rs][cf] = a;
  s2[rs][cf] = b;
  __syncthreads();
  if (rs == 0) {
    float sa = s1[0][cf] + s1[1][cf] + s1[2][cf] + s1[3][cf];
    float sb = s2[0][cf] + s2[1][cf] + s2[2][cf] + s2[3][cf];
    float mean = sa / (float)NROW;
    float var  = sb / (float)NROW - mean * mean;
    stats[col] = mean;
    stats[FDIM + col] = rsqrtf(var + 1e-5f);
  }
}

// ---------------------------------------------------------------------------
// K4: normalize + leaky relu
// ---------------------------------------------------------------------------
__global__ __launch_bounds__(256) void k4_norm(const float* __restrict__ Y,
                                               const float* __restrict__ stats,
                                               float* __restrict__ out) {
  size_t i = (size_t)blockIdx.x * 256 + threadIdx.x;
  int col = (int)(i & (size_t)(FDIM - 1));
  float y = (Y[i] - stats[col]) * stats[FDIM + col];
  out[i] = (y >= 0.0f) ? y : 0.01f * y;
}

// ---------------------------------------------------------------------------
extern "C" void kernel_launch(void* const* d_in, const int* in_sizes, int n_in,
                              void* d_out, int out_size, void* d_ws, size_t ws_size,
                              hipStream_t stream) {
  const float* X  = (const float*)d_in[0];
  // d_in[1] = adj (unused by the module)
  const float* W  = (const float*)d_in[2];
  const float* aw = (const float*)d_in[3];
  const float* ab = (const float*)d_in[4];
  float* out = (float*)d_out;

  char* ws = (char*)d_ws;
  bf16_t* Whb = (bf16_t*)(ws);                      // 8 MB  [NROW,FDIM]
  bf16_t* WhT = (bf16_t*)(ws + ((size_t)8 << 20));  // 8 MB  [FDIM,NROW]
  bf16_t* Tb  = (bf16_t*)(ws + ((size_t)16 << 20)); // 8 MB  [NROW,FDIM]
  float*  Y   = (float*)(ws + ((size_t)24 << 20));  // 16 MB [NROW,FDIM]
  float*  st  = (float*)(ws + ((size_t)40 << 20));  // 4 KB

  (void)hipFuncSetAttribute(reinterpret_cast<const void*>(k2_attention),
                            hipFuncAttributeMaxDynamicSharedMemorySize,
                            (int)K2_SMEM_BYTES);

  k1_gemm_tanh<<<dim3(NROW / 128, FDIM / 64), dim3(256), 0, stream>>>(
      X, W, aw, ab, Whb, WhT, Tb);
  k2_attention<<<dim3(NROW / 64), dim3(256), K2_SMEM_BYTES, stream>>>(
      Whb, WhT, Tb, X, Y);
  k3_bnstats<<<dim3(FDIM / 64), dim3(256), 0, stream>>>(Y, st);
  k4_norm<<<dim3((NROW * FDIM) / 256), dim3(256), 0, stream>>>(Y, st, out);
}
